// LengthTransform_6820408066383
// MI455X (gfx1250) — compile-verified, matrix operand order checked by reading, not run on hardware
//
#include <hip/hip_runtime.h>
#include <hip/hip_bf16.h>

typedef __attribute__((ext_vector_type(16))) _Float16 v16h;
typedef __attribute__((ext_vector_type(8)))  float    v8f;

// ---------------------------------------------------------------------------
// Kernel 1: ratio[b] = sum(mask[b,:]) / tgt_lens[b]
// ---------------------------------------------------------------------------
__global__ void lt_ratio(const float* __restrict__ mask,
                         const int* __restrict__ tgt_lens,
                         float* __restrict__ ratio, int S) {
  __shared__ float red[256];
  const int b   = blockIdx.x;
  const int tid = threadIdx.x;
  float s = 0.f;
  for (int i = tid; i < S; i += 256) s += mask[(size_t)b * S + i];
  red[tid] = s;
  __syncthreads();
  for (int w = 128; w > 0; w >>= 1) {
    if (tid < w) red[tid] += red[tid + w];
    __syncthreads();
  }
  if (tid == 0) ratio[b] = red[0] / (float)tgt_lens[b];
}

// ---------------------------------------------------------------------------
// Kernel 2: pre-swizzle X*mask into f16 WMMA B-matrix layout (D assumed 64).
// Tile (b, chunk c, dtile dt) = 32 lanes x v16h. 16-bit B 32x16 layout:
//   lane L: n = L&15, hi = L>>4; element e (vgpr vv=e>>1, half lo=e&1):
//   K = 2*(vv&3) + lo + 8*hi + 16*(vv>>2)
// ---------------------------------------------------------------------------
__global__ void lt_prep_x(const float* __restrict__ x,
                          const float* __restrict__ mask,
                          v16h* __restrict__ xswz, int S, int SC, int total) {
  const int gid = blockIdx.x * blockDim.x + threadIdx.x;
  if (gid >= total) return;
  const int L  = gid & 31;
  int rest     = gid >> 5;
  const int dt = rest & 3; rest >>= 2;
  const int c  = rest % SC;
  const int b  = rest / SC;
  const int hi = L >> 4, n = L & 15;
  const float* xb = x    + (size_t)b * S * 64;
  const float* mb = mask + (size_t)b * S;
  v16h v;
#pragma unroll
  for (int e = 0; e < 16; ++e) {
    const int vv = e >> 1, lo = e & 1;
    const int K  = 2 * (vv & 3) + lo + (hi << 3) + ((vv >> 2) << 4);
    const int s  = (c << 5) + K;
    v[e] = (_Float16)(xb[(size_t)s * 64 + dt * 16 + n] * mb[s]);
  }
  xswz[gid] = v;
}

// Mask tiles: B[k][n] = mask[s0+k] for all n (for denominator WMMA)
__global__ void lt_prep_m(const float* __restrict__ mask,
                          v16h* __restrict__ mswz, int S, int SC, int total) {
  const int gid = blockIdx.x * blockDim.x + threadIdx.x;
  if (gid >= total) return;
  const int L  = gid & 31;
  const int c  = (gid >> 5) % SC;
  const int b  = (gid >> 5) / SC;
  const int hi = L >> 4;
  const float* mb = mask + (size_t)b * S;
  v16h v;
#pragma unroll
  for (int e = 0; e < 16; ++e) {
    const int vv = e >> 1, lo = e & 1;
    const int K  = 2 * (vv & 3) + lo + (hi << 3) + ((vv >> 2) << 4);
    v[e] = (_Float16)mb[(c << 5) + K];
  }
  mswz[gid] = v;
}

// ---------------------------------------------------------------------------
// Kernel 3: main. One wave (32 lanes) per 16-target-column tile.
// D hardcoded to 64 (4 N-tiles).  A 16x32 f16 layout:
//   lane L: m = L&15 (row t), hi = L>>4; half h: K = h + (h&8) + 8*hi
// ---------------------------------------------------------------------------
__global__ __launch_bounds__(128, 1)
void lt_main(const v16h* __restrict__ xswz, const v16h* __restrict__ mswz,
             const float* __restrict__ mask, const float* __restrict__ ratio,
             const float* __restrict__ lsp, float* __restrict__ out,
             int S, int T, int SC, int totalTiles) {
  const int wave = threadIdx.x >> 5;
  const int lane = threadIdx.x & 31;
  const int tile = blockIdx.x * 4 + wave;
  if (tile >= totalTiles) return;
  const int tilesPerB = T >> 4;
  const int b  = tile / tilesPerB;
  const int t0 = (tile % tilesPerB) << 4;
  const int m16 = lane & 15, hi = lane >> 4;

  const float ls   = lsp[0];
  const float coef = 0.5f / (ls * ls);
  const float a    = ratio[b] * (float)(t0 + m16);
  const float* mrow = mask + (size_t)b * S;

  // ---- pass 1: column max of masked logits (lane pair splits S) ----
  float mx = -3.0e38f;
  const int half = S >> 1;
  const int sBeg = hi * half;
  for (int i = 0; i < half; i += 4) {
    const float4 m4 = *(const float4*)(mrow + sBeg + i);
#pragma unroll
    for (int j = 0; j < 4; ++j) {
      const float msk = j == 0 ? m4.x : j == 1 ? m4.y : j == 2 ? m4.z : m4.w;
      const float d   = (float)(sBeg + i + j) - a;
      const float lgB = fmaf(d * d, -coef, 1e10f);      // logit + 1e10
      mx = fmaxf(mx, fmaf(msk, lgB, -1e10f));           // msk*logit-(1-msk)*1e10
    }
  }
  mx = fmaxf(mx, __shfl_xor(mx, 16));

  // ---- main streaming loop: 5 WMMAs per 32-wide s-chunk ----
  v8f acc0 = {}, acc1 = {}, acc2 = {}, acc3 = {}, accm = {};
  const v16h* bbase = xswz + (size_t)b * SC * 128 + lane;
  const v16h* mbase = mswz + (size_t)b * SC * 32  + lane;
  for (int c = 0; c < SC; ++c) {
    const float sb = (float)(c << 5);
    v16h A;
#pragma unroll
    for (int h = 0; h < 16; ++h) {
      const int K   = h + (h & 8) + (hi << 3);
      const float d = (sb + (float)K) - a;
      float arg = fmaf(d * d, -coef, -mx);
      arg = fminf(arg, 0.f);                 // f16-safe: weights in (0,1]
      A[h] = (_Float16)__expf(arg);
    }
    const v16h* bp = bbase + (size_t)c * 128;
    const v16h B0 = bp[0], B1 = bp[32], B2 = bp[64], B3 = bp[96];
    const v16h BM = mbase[(size_t)c * 32];
    if (c + 1 < SC) {
      // locality 3 -> WGP-scope prefetch: pull next chunk into near caches
      __builtin_prefetch((const void*)(bp + 128), 0, 3);   // global_prefetch_b8
      __builtin_prefetch((const void*)(mbase + (size_t)(c + 1) * 32), 0, 3);
    }
    acc0 = __builtin_amdgcn_wmma_f32_16x16x32_f16(false, A, false, B0, (short)0, acc0, false, false);
    acc1 = __builtin_amdgcn_wmma_f32_16x16x32_f16(false, A, false, B1, (short)0, acc1, false, false);
    acc2 = __builtin_amdgcn_wmma_f32_16x16x32_f16(false, A, false, B2, (short)0, acc2, false, false);
    acc3 = __builtin_amdgcn_wmma_f32_16x16x32_f16(false, A, false, BM == BM ? B3 : B3, (short)0, acc3, false, false);
    accm = __builtin_amdgcn_wmma_f32_16x16x32_f16(false, A, false, BM, (short)0, accm, false, false);
  }

  // ---- epilogue: rows of C map to regs; denominator sits in the same regs ----
#pragma unroll
  for (int rr = 0; rr < 8; ++rr) {
    const int M = rr + (hi << 3);                    // matrix row (local t)
    const float sc = __builtin_amdgcn_rcpf(accm[rr]); // v_rcp_f32 (fast)
    const size_t off = ((size_t)b * T + (t0 + M)) * 64 + m16;
    out[off]      = acc0[rr] * sc;
    out[off + 16] = acc1[rr] * sc;
    out[off + 32] = acc2[rr] * sc;
    out[off + 48] = acc3[rr] * sc;
  }
}

// ---------------------------------------------------------------------------
// Kernel 4: tgt_mask[b,t] = (t < tgt_lens[b]) as 0/1
// ---------------------------------------------------------------------------
__global__ void lt_mask(const int* __restrict__ tgt_lens,
                        float* __restrict__ om, int T, int total) {
  const int i = blockIdx.x * blockDim.x + threadIdx.x;
  if (i >= total) return;
  om[i] = ((i % T) < tgt_lens[i / T]) ? 1.f : 0.f;
}

extern "C" void kernel_launch(void* const* d_in, const int* in_sizes, int n_in,
                              void* d_out, int out_size, void* d_ws, size_t ws_size,
                              hipStream_t stream) {
  const float* x    = (const float*)d_in[0];
  const float* mask = (const float*)d_in[1];
  const float* lsp  = (const float*)d_in[2];
  const int*   tlen = (const int*)d_in[3];
  float* out = (float*)d_out;

  const int B = in_sizes[3];
  const int S = in_sizes[1] / B;
  const int D = in_sizes[0] / in_sizes[1];      // = 64 (layout assumes this)
  const int T = out_size / (B * (D + 1));
  const int SC = S / 32;

  // workspace: [ratios | xswz f16 tiles | mask f16 tiles]
  float* ratio = (float*)d_ws;
  v16h*  xswz  = (v16h*)((char*)d_ws + 256);
  v16h*  mswz  = xswz + (size_t)B * SC * 128;

  lt_ratio<<<B, 256, 0, stream>>>(mask, tlen, ratio, S);

  const int px = B * SC * 4 * 32;
  lt_prep_x<<<(px + 255) / 256, 256, 0, stream>>>(x, mask, xswz, S, SC, px);
  const int pm = B * SC * 32;
  lt_prep_m<<<(pm + 255) / 256, 256, 0, stream>>>(mask, mswz, S, SC, pm);

  const int tiles = B * (T / 16);
  lt_main<<<(tiles + 3) / 4, 128, 0, stream>>>(xswz, mswz, mask, ratio, lsp,
                                               out, S, T, SC, tiles);

  const int mt = B * T;
  lt_mask<<<(mt + 255) / 256, 256, 0, stream>>>(tlen, out + (size_t)B * T * D, T, mt);
}
